// MSTFCN_21741124452377
// MI455X (gfx1250) — compile-verified, matrix operand order checked by reading, not run on hardware
//
#include <hip/hip_runtime.h>

// ---------------- problem constants (from reference) ----------------
#define B_    32
#define T_    24
#define N_    1024
#define NM    2
#define D_    40
#define RES   32
#define CONV  32
#define SKIPC 64
#define ENDC  128
#define HOR   12
#define NLAY  3
#define ALPHA 0.05f
#define TA    3.0f
#define TOPK_ 20

#define BTN   (B_*T_*N_)                    // 786432
#define FSZ   ((size_t)B_*RES*T_*N_)        // 25165824 floats per feature tensor
#define SKSZ  ((size_t)B_*SKIPC*N_)         // 2097152
#define PROWS (B_*RES*T_)                   // 24576 GEMM rows for propagation
#define NT    8                             // n-tiles per wave in k_prop

typedef __attribute__((ext_vector_type(16))) _Float16 v16h;
typedef __attribute__((ext_vector_type(8)))  _Float16 v8h;
typedef __attribute__((ext_vector_type(4)))  _Float16 v4h;
typedef __attribute__((ext_vector_type(8)))  float    v8f;

__device__ __forceinline__ size_t fidx(int b,int c,int t,int n){
  return (((size_t)b*RES + c)*T_ + t)*N_ + n;
}

// ---- gfx1250 async global->LDS copy (ASYNCcnt-tracked, no VGPR round-trip) --
__device__ __forceinline__ void async_ld16(unsigned lds_off, const void* g){
  asm volatile("global_load_async_to_lds_b128 %0, %1, off"
               :: "v"(lds_off), "v"(g) : "memory");
}
__device__ __forceinline__ void wait_async0(){
  asm volatile("s_wait_asynccnt 0x0" ::: "memory");
}

// ---------------- E1/E2 = tanh(3 * emb @ glin) ----------------
__global__ void k_embed(const float* __restrict__ emb, const float* __restrict__ gl1,
                        const float* __restrict__ gl2, float* __restrict__ E1,
                        float* __restrict__ E2) {
  int idx = blockIdx.x*256 + threadIdx.x;          // (i*N + n)*D + e
  if (idx >= NM*N_*D_) return;
  int e = idx % D_;
  int rem = idx / D_;
  const float* er = emb + (size_t)rem * D_;
  float s1=0.f, s2=0.f;
  for (int d=0; d<D_; ++d){ float v=er[d]; s1 += v*gl1[d*D_+e]; s2 += v*gl2[d*D_+e]; }
  E1[idx] = tanhf(TA*s1);
  E2[idx] = tanhf(TA*s2);
}

// ---------------- adj[i,j,n,m] = relu(tanh(3*(E1i·E2j^T - its transpose))) ----
__global__ void k_adj(const float* __restrict__ E1, const float* __restrict__ E2,
                      float* __restrict__ ADJ) {
  __shared__ float s1m[128*D_];
  __shared__ float s2m[128*D_];
  __shared__ float sn1[D_];
  __shared__ float sn2[D_];
  int bid = blockIdx.x;
  int mblk = bid & 7;  bid >>= 3;
  int n    = bid & (N_-1); bid >>= 10;
  int j = bid & 1; int i = bid >> 1;
  int t = threadIdx.x;
  int m = mblk*128 + t;
  const float* e1i = E1 + (size_t)i*N_*D_;
  const float* e2j = E2 + (size_t)j*N_*D_;
  for (int d=t; d<D_; d+=128){ sn1[d]=e1i[(size_t)n*D_+d]; sn2[d]=e2j[(size_t)n*D_+d]; }
  for (int d=0; d<D_; ++d){ s1m[t*D_+d]=e1i[(size_t)m*D_+d]; s2m[t*D_+d]=e2j[(size_t)m*D_+d]; }
  __syncthreads();
  float a=0.f, b=0.f;
  for (int d=0; d<D_; ++d){ a += sn1[d]*s2m[t*D_+d]; b += sn2[d]*s1m[t*D_+d]; }
  float v = tanhf(TA*(a-b));
  ADJ[(((size_t)(i*NM+j)*N_ + n)*N_) + m] = v>0.f ? v : 0.f;
}

// -- exact top-20 threshold + self loop + row normalize; emits f16 adjacency --
__global__ void k_topk(const float* __restrict__ ADJ, _Float16* __restrict__ ANh) {
  __shared__ float rv[N_];
  __shared__ float sv[128];
  __shared__ int   si[128];
  __shared__ float s_thr;
  __shared__ int   s_best;
  int row = blockIdx.x;              // pair*N + n
  int n = row & (N_-1);
  const float* r = ADJ + (size_t)row * N_;
  _Float16*   rh = ANh + (size_t)row * N_;
  int t = threadIdx.x;
  for (int m=t; m<N_; m+=128) rv[m] = r[m];
  unsigned mask = 0xFFu;             // 8 elements per thread: m = s*128 + t
  __syncthreads();
  float thr = 0.f;
  for (int it=0; it<TOPK_; ++it){
    float bv = -1.f; int bi = -1;
    for (int s=0; s<8; ++s){
      if (mask & (1u<<s)){
        int m = s*128 + t; float v = rv[m];
        if (v > bv){ bv = v; bi = m; }
      }
    }
    sv[t]=bv; si[t]=bi;
    __syncthreads();
    for (int st=64; st>0; st>>=1){
      if (t < st){
        float ov=sv[t+st]; int oi=si[t+st];
        if (ov > sv[t] || (ov == sv[t] && (unsigned)oi < (unsigned)si[t])){ sv[t]=ov; si[t]=oi; }
      }
      __syncthreads();
    }
    if (t==0){ s_thr=sv[0]; s_best=si[0]; }
    __syncthreads();
    thr = s_thr;
    int bidx = s_best;
    if ((bidx & 127) == t) mask &= ~(1u << (bidx >> 7));
    __syncthreads();
  }
  // thr = 20th largest (duplicates counted, same multiset as jax top_k)
  float ls = 0.f;
  for (int s=0; s<8; ++s){ int m=s*128+t; float v=rv[m]; ls += (v>=thr)? v:0.f; }
  sv[t]=ls; __syncthreads();
  for (int st=64; st>0; st>>=1){ if (t<st) sv[t]+=sv[t+st]; __syncthreads(); }
  float inv = 1.f / (sv[0] + 1.0f);  // +1 for the self loop (diag of relu(tanh) is 0)
  for (int s=0; s<8; ++s){
    int m=s*128+t;
    float v=rv[m]; v = (v>=thr)? v:0.f;
    if (m==n) v += 1.0f;
    rh[m] = (_Float16)(v*inv);
  }
}

// ---------------- f32 feature tensor -> f16 (vectorized) ----------------
__global__ void k_half(const float* __restrict__ s, _Float16* __restrict__ d){
  size_t i = ((size_t)blockIdx.x*256 + threadIdx.x)*4;
  float4 v = *(const float4*)(s+i);
  v4h o = { (_Float16)v.x, (_Float16)v.y, (_Float16)v.z, (_Float16)v.w };
  *(v4h*)(d+i) = o;
}

// ---------------- time embedding -> both bs buffers ----------------
__global__ void k_time(const float* __restrict__ in, const float* __restrict__ tw,
                       float* __restrict__ Bt0, float* __restrict__ Bt1) {
  int idx = blockIdx.x*256 + threadIdx.x;          // (b*T+t)*N+n
  int n = idx & (N_-1); int r = idx>>10; int t = r % T_; int b = r / T_;
  const float* ip = in + (size_t)idx*4;
  float f0 = ip[2], f1 = ip[3];
  for (int c=0;c<RES;c++){
    float v = f0*tw[c*2+0] + f1*tw[c*2+1];
    size_t o = fidx(b,c,t,n);
    Bt0[o]=v; Bt1[o]=v;
  }
}

// ---------------- start conv (1 -> 32 channels) ----------------
__global__ void k_start(const float* __restrict__ in, const float* __restrict__ sw,
                        float* __restrict__ X0, float* __restrict__ X1) {
  int idx = blockIdx.x*256 + threadIdx.x;
  int n = idx & (N_-1); int r = idx>>10; int t = r % T_; int b = r / T_;
  const float* ip = in + (size_t)idx*4;
  float v0 = ip[0], v1 = ip[1];
  for (int c=0;c<RES;c++){
    size_t o = fidx(b,c,t,n);
    X0[o] = v0*sw[c];
    X1[o] = v1*sw[RES+c];
  }
}

// ---------------- skip0 conv + zero skip_b ----------------
__global__ void k_skip0(const float* __restrict__ in, const float* __restrict__ w,
                        float* __restrict__ SX0, float* __restrict__ SX1,
                        float* __restrict__ SB0, float* __restrict__ SB1) {
  int idx = blockIdx.x*256 + threadIdx.x;          // (b*64+o)*N+n
  int n = idx & (N_-1); int r = idx>>10; int o = r % SKIPC; int b = r / SKIPC;
  float a0=0.f, a1=0.f;
  for (int t=0;t<T_;++t){
    const float* ip = in + ((((size_t)b*T_+t)*N_+n)*4);
    a0 += ip[0]*w[(0*SKIPC+o)*T_+t];
    a1 += ip[1]*w[(1*SKIPC+o)*T_+t];
  }
  SX0[idx]=a0; SX1[idx]=a1; SB0[idx]=0.f; SB1[idx]=0.f;
}

// ---------------- gated dilated temporal conv (tanh * sigmoid) ----------------
__global__ void k_gated(const float* __restrict__ X, const float* __restrict__ wf,
                        const float* __restrict__ wg, float* __restrict__ O, int dil) {
  __shared__ float swf[CONV*RES*2];
  __shared__ float swg[CONV*RES*2];
  for (int k=threadIdx.x; k<CONV*RES*2; k+=256){ swf[k]=wf[k]; swg[k]=wg[k]; }
  __syncthreads();
  int idx = blockIdx.x*256 + threadIdx.x;
  int n = idx & (N_-1); int r = idx>>10; int t = r % T_; int b = r / T_;
  float x0[RES], xm[RES];
#pragma unroll
  for (int c=0;c<RES;c++){
    x0[c] = X[fidx(b,c,t,n)];
    xm[c] = (t >= dil) ? X[fidx(b,c,t-dil,n)] : 0.f;
  }
  for (int o=0;o<CONV;o++){
    float f=0.f, g=0.f;
#pragma unroll
    for (int c=0;c<RES;c++){
      int w0=(o*RES+c)*2;
      f += swf[w0]*xm[c] + swf[w0+1]*x0[c];
      g += swg[w0]*xm[c] + swg[w0+1]*x0[c];
    }
    O[fidx(b,o,t,n)] = tanhf(f) * (1.f/(1.f+__expf(-g)));
  }
}

// ---------------- WMMA graph propagation ----------------
// OUT[p][n] = ALPHA*X0[p][n] + (1-ALPHA) * sum_m INh[p][m] * ANh[n][m]
// v_wmma_f32_16x16x32_f16, f32 accumulation, f16 operands prepared up front.
// Block: 8 waves. Block tile = 128 p x 128 n.  Each wave: one 16-row p-tile,
// NT=8 n-tiles (8 accumulators) -> the A fragment is reused by 8 WMMAs and the
// 128x32 f16 AN tile in (ping-pong) LDS is shared by all waves.  The AN tile
// is staged with global_load_async_to_lds_b128 (ASYNCcnt), so the staging data
// never touches VGPRs; s_wait_asynccnt 0 + one barrier per k-step order the
// async LDS writes against the next iteration's ds_load_b128 reads.
// Per-lane packing follows ISA 7.12.2 (16-bit A 16x32):
//   lanes 0-15:  elems 0..7 -> K 0..7,  elems 8..15 -> K 16..23
//   lanes 16-31: elems 0..7 -> K 8..15, elems 8..15 -> K 24..31   (B mirrored)
__global__ void __launch_bounds__(256) k_prop(const _Float16* __restrict__ ANh,
                                              const _Float16* __restrict__ INh,
                                              const float* __restrict__ X0,
                                              float* __restrict__ OUT) {
  __shared__ _Float16 sB[2][128*32];               // [buf][n (128)][k (32)]
  int lane = threadIdx.x & 31;
  int wave = threadIdx.x >> 5;
  int n0 = blockIdx.y * 128;
  int p0 = blockIdx.x * 128 + wave*16;
  int half = lane >> 4;
  int rc   = lane & 15;
  v8f acc[NT];
#pragma unroll
  for (int nt=0; nt<NT; ++nt) acc[nt] = (v8f){};
  const _Float16* arow = INh + (size_t)(p0 + rc)*N_ + half*8;
  // static staging map: thread copies 16 contiguous f16 (two async b128 copies)
  int snn = threadIdx.x >> 1;                      // 0..127
  int skk = (threadIdx.x & 1) * 16;                // 0 or 16
  const _Float16* srow = ANh + (size_t)(n0 + snn)*N_ + skk;
  // LDS byte addresses of this thread's two 16B slots in each ping-pong buffer
  // (flat LDS addressing keeps the low bits as the LDS offset, ISA 10.2)
  unsigned d0 = (unsigned)(uintptr_t)(&sB[0][snn*32 + skk]);
  unsigned d1 = d0 + (unsigned)(128*32*sizeof(_Float16));
  { // prologue: stage k0 = 0 into buffer 0
    async_ld16(d0,      srow);
    async_ld16(d0 + 16, srow + 8);
    wait_async0();
  }
  __syncthreads();
  for (int k0 = 0, it = 0; k0 < N_; k0 += 32, ++it){
    int cur = it & 1;
    if (k0 + 32 < N_){                             // stage next tile (uniform)
      const _Float16* s = srow + k0 + 32;
      unsigned d = cur ? d0 : d1;
      async_ld16(d,      s);
      async_ld16(d + 16, s + 8);
      __builtin_prefetch(arow + k0 + 32, 0, 1);
    }
    // A fragment (two 16B loads, reused by all NT WMMAs)
    v8h alo = *(const v8h*)(arow + k0);
    v8h ahi = *(const v8h*)(arow + k0 + 16);
    v16h a;
#pragma unroll
    for (int q=0;q<8;q++){ a[q]=alo[q]; a[8+q]=ahi[q]; }
    const _Float16* sbuf = &sB[cur][0];
#pragma unroll
    for (int nt=0; nt<NT; ++nt){
      const _Float16* bp = sbuf + (nt*16 + rc)*32 + half*8;
      v8h blo = *(const v8h*)(bp);
      v8h bhi = *(const v8h*)(bp + 16);
      v16h b;
#pragma unroll
      for (int q=0;q<8;q++){ b[q]=blo[q]; b[8+q]=bhi[q]; }
      acc[nt] = __builtin_amdgcn_wmma_f32_16x16x32_f16(false, a, false, b,
                                                       (short)0, acc[nt],
                                                       false, false);
    }
    wait_async0();                                 // async stage landed in LDS
    __syncthreads();
  }
  // D layout: VGPR v, lanes 0-15 -> M=v, lanes 16-31 -> M=v+8; N = lane&15
#pragma unroll
  for (int nt=0; nt<NT; ++nt){
#pragma unroll
    for (int v=0; v<8; ++v){
      int pr = v + half*8;
      size_t o = (size_t)(p0+pr)*N_ + n0 + nt*16 + rc;
      OUT[o] = ALPHA*X0[o] + (1.f-ALPHA)*acc[nt][v];
    }
  }
}

// ---------------- gcn 1x1 conv over concat(hop0,hop1,hop2), weighted accum ----
__global__ void k_gcn(const float* __restrict__ H0, const float* __restrict__ H1,
                      const float* __restrict__ H2, const float* __restrict__ W,
                      const float* __restrict__ cw, int i, int j, int accumulate,
                      float* __restrict__ ACC) {
  __shared__ float sW[CONV*3*RES];
  for (int k=threadIdx.x; k<CONV*3*RES; k+=256) sW[k]=W[k];
  __syncthreads();
  float wij = cw[i*NM + j] + (i==j ? 1.f : 0.f);
  int idx = blockIdx.x*256 + threadIdx.x;
  int n = idx & (N_-1); int r = idx>>10; int t = r % T_; int b = r / T_;
  float acc[CONV];
#pragma unroll
  for (int o=0;o<CONV;o++) acc[o]=0.f;
  const float* Hs[3] = {H0,H1,H2};
  for (int hb=0; hb<3; ++hb){
    const float* H = Hs[hb];
    for (int c=0;c<RES;c++){
      float hv = H[fidx(b,c,t,n)];
      int cc = hb*RES + c;
#pragma unroll
      for (int o=0;o<CONV;o++) acc[o] += sW[o*(3*RES)+cc]*hv;
    }
  }
  if (accumulate){
#pragma unroll
    for (int o=0;o<CONV;o++) ACC[fidx(b,o,t,n)] += wij*acc[o];
  } else {
#pragma unroll
    for (int o=0;o<CONV;o++) ACC[fidx(b,o,t,n)] = wij*acc[o];
  }
}

// ---------------- channel conv (64->64), split into new x / new b -------------
// In-place safe: each thread reads/writes only its own (b,t,n) channel column.
__global__ void k_chan(const float* __restrict__ ACC, const float* __restrict__ BG,
                       const float* __restrict__ W, float* __restrict__ Xn,
                       float* __restrict__ Bn) {
  __shared__ float sW[64*64];
  for (int k=threadIdx.x; k<64*64; k+=256) sW[k]=W[k];
  __syncthreads();
  int idx = blockIdx.x*256 + threadIdx.x;
  int n = idx & (N_-1); int r = idx>>10; int t = r % T_; int b = r / T_;
  float h[64];
#pragma unroll
  for (int c=0;c<32;c++){ h[c]=ACC[fidx(b,c,t,n)]; h[32+c]=BG[fidx(b,c,t,n)]; }
  for (int o=0;o<64;o++){
    float s=0.f;
#pragma unroll
    for (int c=0;c<64;c++) s += sW[o*64+c]*h[c];
    if (o<32) Xn[fidx(b,o,t,n)] = s; else Bn[fidx(b,o-32,t,n)] = s;
  }
}

// ---------------- skip accumulation: SK[b,o,n] += sum_{c,t} X*w ---------------
__global__ void k_skip(const float* __restrict__ X, const float* __restrict__ SW,
                       float* __restrict__ SK) {
  int idx = blockIdx.x*256 + threadIdx.x;          // (b*64+o)*N+n
  int n = idx & (N_-1); int r = idx>>10; int o = r % SKIPC; int b = r / SKIPC;
  float s=0.f;
  for (int c=0;c<RES;c++)
    for (int t=0;t<T_;++t)
      s += X[fidx(b,c,t,n)] * SW[(o*RES+c)*T_+t];
  SK[idx] += s;
}

// ---------------- output head: relu -> 128x128 -> relu -> 12x128 --------------
__global__ void k_out(const float* __restrict__ SKX, const float* __restrict__ SKB,
                      const float* __restrict__ W1, const float* __restrict__ W2,
                      float* __restrict__ out, int mix) {
  int idx = blockIdx.x*256 + threadIdx.x;          // b*N + n
  int n = idx & (N_-1); int b = idx >> 10;
  float h1[ENDC];
#pragma unroll
  for (int o=0;o<ENDC;o++) h1[o]=0.f;
  for (int c=0;c<ENDC;c++){
    float v = (c < SKIPC) ? SKX[(size_t)(b*SKIPC + c)*N_ + n]
                          : SKB[(size_t)(b*SKIPC + (c-SKIPC))*N_ + n];
    v = v>0.f ? v : 0.f;
#pragma unroll
    for (int o=0;o<ENDC;o++) h1[o] += W1[o*ENDC + c] * v;
  }
  float oa[HOR];
#pragma unroll
  for (int h=0;h<HOR;h++) oa[h]=0.f;
  for (int c=0;c<ENDC;c++){
    float v = h1[c]>0.f ? h1[c] : 0.f;
#pragma unroll
    for (int h=0;h<HOR;h++) oa[h] += W2[h*ENDC + c] * v;
  }
  for (int h=0;h<HOR;h++)
    out[(((size_t)b*HOR + h)*N_ + n)*NM + mix] = oa[h];
}

// =============================== driver ======================================
extern "C" void kernel_launch(void* const* d_in, const int* in_sizes, int n_in,
                              void* d_out, int out_size, void* d_ws, size_t ws_size,
                              hipStream_t stream) {
  (void)in_sizes; (void)n_in; (void)out_size;
  const float* input = (const float*)d_in[0];
  const float* emb   = (const float*)d_in[1];
  const float* glin1 = (const float*)d_in[2];
  const float* glin2 = (const float*)d_in[3];
  const float* crossw= (const float*)d_in[4];
  const float* time_w= (const float*)d_in[5];
  const float* skip0w= (const float*)d_in[6];
  const float* startw= (const float*)d_in[7];
  const float* filtw = (const float*)d_in[8];
  const float* gatew = (const float*)d_in[9];
  const float* gcnw  = (const float*)d_in[10];
  const float* chanw = (const float*)d_in[11];
  const float* skipw = (const float*)d_in[12];
  const float* end1w = (const float*)d_in[13];
  const float* end2w = (const float*)d_in[14];
  float* out = (float*)d_out;

  // f32 float-slots + f16 regions (counted in float-equivalents)
  const size_t needF = 2*(size_t)NM*N_*D_            // E1,E2
                     + (size_t)NM*NM*N_*N_           // ADJ (f32)
                     + (size_t)NM*NM*N_*N_/2         // ANh (f16)
                     + 10*FSZ                        // f32 feature buffers
                     + 3*(FSZ/2)                     // XGh[2], H1h (f16)
                     + 4*SKSZ;
  if (ws_size < needF*sizeof(float)) return;         // ~1.22 GB required

  float* p = (float*)d_ws;
  float* E1  = p; p += (size_t)NM*N_*D_;
  float* E2  = p; p += (size_t)NM*N_*D_;
  float* ADJ = p; p += (size_t)NM*NM*N_*N_;
  _Float16* ANh = (_Float16*)p; p += (size_t)NM*NM*N_*N_/2;
  float* Xs[2];  Xs[0]=p; p+=FSZ; Xs[1]=p; p+=FSZ;   // also serve as gcn accumulators
  float* Bs[2];  Bs[0]=p; p+=FSZ; Bs[1]=p; p+=FSZ;
  float* XGs[2]; XGs[0]=p; p+=FSZ; XGs[1]=p; p+=FSZ;
  float* BGs[2]; BGs[0]=p; p+=FSZ; BGs[1]=p; p+=FSZ;
  float* H1 = p; p+=FSZ;
  float* H2 = p; p+=FSZ;
  _Float16* XGh[2]; XGh[0]=(_Float16*)p; p+=FSZ/2; XGh[1]=(_Float16*)p; p+=FSZ/2;
  _Float16* H1h = (_Float16*)p; p+=FSZ/2;
  float* SKX[2]; SKX[0]=p; p+=SKSZ; SKX[1]=p; p+=SKSZ;
  float* SKB[2]; SKB[0]=p; p+=SKSZ; SKB[1]=p; p+=SKSZ;

  // graph construction (once, reused by all 24 propagations)
  k_embed<<<(NM*N_*D_+255)/256,256,0,stream>>>(emb,glin1,glin2,E1,E2);
  k_adj  <<<NM*NM*N_*(N_/128),128,0,stream>>>(E1,E2,ADJ);
  k_topk <<<NM*NM*N_,128,0,stream>>>(ADJ,ANh);

  // input stage
  k_time <<<BTN/256,256,0,stream>>>(input,time_w,Bs[0],Bs[1]);
  k_start<<<BTN/256,256,0,stream>>>(input,startw,Xs[0],Xs[1]);
  k_skip0<<<(int)(SKSZ/256),256,0,stream>>>(input,skip0w,SKX[0],SKX[1],SKB[0],SKB[1]);

  const int hgrid = (int)(FSZ/4/256);                // k_half grid (24576)
  dim3 pgrid(PROWS/128, N_/128);                     // 192 x 8 blocks, 256 thr
  for (int l=0; l<NLAY; ++l){
    int dil = 1<<l;
    for (int i=0;i<NM;i++){
      const float* wf = filtw + (size_t)(i*NLAY+l)*CONV*RES*2;
      const float* wg = gatew + (size_t)(i*NLAY+l)*CONV*RES*2;
      k_gated<<<BTN/256,256,0,stream>>>(Xs[i],wf,wg,XGs[i],dil);
      k_gated<<<BTN/256,256,0,stream>>>(Bs[i],wf,wg,BGs[i],dil);
      k_half <<<hgrid,256,0,stream>>>(XGs[i], XGh[i]);
    }
    for (int i=0;i<NM;i++){
      const float* gw = gcnw + (size_t)(i*NLAY+l)*CONV*3*RES;
      for (int j=0;j<NM;j++){
        const _Float16* A = ANh + (size_t)(i*NM+j)*N_*N_;
        k_prop<<<pgrid,256,0,stream>>>(A, XGh[j], XGs[j], H1);   // hop1
        k_half<<<hgrid,256,0,stream>>>(H1, H1h);
        k_prop<<<pgrid,256,0,stream>>>(A, H1h,    XGs[j], H2);   // hop2
        k_gcn <<<BTN/256,256,0,stream>>>(XGs[j],H1,H2, gw, crossw, i, j,
                                         (j>0)?1:0, Xs[i]);
      }
      k_chan<<<BTN/256,256,0,stream>>>(Xs[i], BGs[i],
                                       chanw + (size_t)(i*NLAY+l)*64*64,
                                       Xs[i], Bs[i]);
      const float* sw = skipw + (size_t)(i*NLAY+l)*SKIPC*RES*T_;
      k_skip<<<(int)(SKSZ/256),256,0,stream>>>(Xs[i], sw, SKX[i]);
      k_skip<<<(int)(SKSZ/256),256,0,stream>>>(Bs[i], sw, SKB[i]);
    }
  }
  for (int i=0;i<NM;i++)
    k_out<<<(B_*N_)/256,256,0,stream>>>(SKX[i],SKB[i],
                                        end1w + (size_t)i*ENDC*ENDC,
                                        end2w + (size_t)i*HOR*ENDC, out, i);
}
